// CycleMatcher_28363964023395
// MI455X (gfx1250) — matrix operand order repaired and checked by reading, not compiled, not run on hardware
//
#include <hip/hip_runtime.h>
#include <hip/hip_bf16.h>

typedef __attribute__((ext_vector_type(16))) _Float16 v16h;
typedef __attribute__((ext_vector_type(2)))  _Float16 v2h;
typedef __attribute__((ext_vector_type(8)))  float    v8f;

#define BQ 4
#define MQ 8192
#define NQ 8192
#define DQ 64
#define TILE 128
#define PITCH 66     // A-tile LDS row pitch in halves (64 -> 66 breaks bank conflicts)
#define BFP 17       // B-frag lane-block pitch in words (17 coprime 64 => conflict-free)

// 32-bit in-workgroup key: high 25 bits = biased-positive-float bits (value
// order == unsigned order), low 7 bits = (127 - local_idx) so umax implements
// "max value, ties -> smallest index" (jnp.argmin tie-break).  7 dropped
// mantissa bits = 3e-5 granularity, below the f16 input noise (~1.2e-4).
#define KMASK 0xFFFFFF80u

// ---- init: packed bests = 0 (loses to every real candidate), m1=-1, ms1=0 ---
__global__ void matcher_init(unsigned long long* __restrict__ rowbest,
                             unsigned long long* __restrict__ colbest,
                             float* __restrict__ out) {
    int idx = blockIdx.x * blockDim.x + threadIdx.x;
    if (idx >= BQ * MQ) return;
    rowbest[idx] = 0ull;
    colbest[idx] = 0ull;                       // M == N, one index covers both
    const size_t BM = (size_t)BQ * MQ, BN = (size_t)BQ * NQ;
    out[2 * BM + idx]      = -1.0f;            // m1
    out[2 * BM + BN + idx] = 0.0f;             // ms1
}

// ---- fused f16-WMMA GEMM tile + row/col argmax reduction --------------------
__global__ __launch_bounds__(256)
void matcher_gemm_argmax(const float* __restrict__ desc0,
                         const float* __restrict__ desc1,
                         unsigned long long* __restrict__ rowbest,
                         unsigned long long* __restrict__ colbest) {
    __shared__ _Float16 As[TILE * PITCH];
    __shared__ _Float16 Bf[8 * 32 * BFP * 2];   // fragment-major B tile
    __shared__ unsigned ldsRow[TILE];
    __shared__ unsigned ldsCol[TILE];

    const int b    = blockIdx.z;
    const int mt   = blockIdx.y;
    const int nt   = blockIdx.x;
    const int t    = threadIdx.x;
    const int wave = t >> 5;
    const int lane = t & 31;
    const int l15  = lane & 15;
    const int hlf  = lane >> 4;

    if (t < 128) ldsRow[t] = 0u; else ldsCol[t - 128] = 0u;

    // ---- stage fp32 tiles -> f16 in LDS ------------------------------------
    // A: row-major, pitch 66.  B: fragment-major so each (subtile,lane)'s 16
    // WMMA B-frag words are contiguous.  This code's contiguous-k B-layout (32x16, 16-bit):
    //   lane = (n&15) | (((k>>4)&1)<<4),  word v = (k>>5)*8 + (k&15)/2
    const float* gA = desc0 + ((size_t)b * MQ + (size_t)mt * TILE) * DQ;
    const float* gB = desc1 + ((size_t)b * NQ + (size_t)nt * TILE) * DQ;
    for (int i = t; i < TILE * 16; i += 256) {
        int row = i >> 4, c4 = i & 15;
        float4 va = ((const float4*)gA)[row * 16 + c4];
        float4 vb = ((const float4*)gB)[row * 16 + c4];
        int o = row * PITCH + c4 * 4;
        As[o + 0] = (_Float16)va.x; As[o + 1] = (_Float16)va.y;
        As[o + 2] = (_Float16)va.z; As[o + 3] = (_Float16)va.w;

        int k0    = c4 * 4;                       // covers k0..k0+3 (same block)
        int st    = row >> 4;
        int blane = (row & 15) | (((k0 >> 4) & 1) << 4);
        int v0    = ((k0 >> 5) << 3) + ((k0 & 15) >> 1);
        int baseh = ((st * 32 + blane) * BFP + v0) * 2;
        v2h p01; p01.x = (_Float16)vb.x; p01.y = (_Float16)vb.y;
        v2h p23; p23.x = (_Float16)vb.z; p23.y = (_Float16)vb.w;
        *(v2h*)&Bf[baseh]     = p01;              // word v0   : k0,k0+1
        *(v2h*)&Bf[baseh + 2] = p23;              // word v0+1 : k0+2,k0+3
    }
    __syncthreads();

    // ---- A fragments for this wave's 16-row strip (K=0..31 and K=32..63) ----
    // 16-bit A 16x32 layout (contiguous k, matches B frags): lane L -> M=L&15;
    //   VGPR v pair K offset kk = v*2 + (L>>4)*16
    const int mstrip = wave * 16;
    v16h a0, a1;
#pragma unroll
    for (int v = 0; v < 8; ++v) {
        int kk = (v << 1) + (hlf << 4);
        int base = (mstrip + l15) * PITCH + kk;
        v2h p0 = *(const v2h*)&As[base];
        v2h p1 = *(const v2h*)&As[base + 32];
        a0[2 * v] = p0.x; a0[2 * v + 1] = p0.y;
        a1[2 * v] = p1.x; a1[2 * v + 1] = p1.y;
    }

    // Per-lane row-argmax state as u32 keys: acc[r] always belongs to row
    // (mstrip + 8*hlf + r); streaming umax across the 8 N-subtiles.
    unsigned rbest[8];
#pragma unroll
    for (int r = 0; r < 8; ++r) rbest[r] = 0u;

    const unsigned e7mBase = 127u - (unsigned)(mstrip + hlf * 8);  // col keys
    const unsigned e7nBase = 127u - (unsigned)l15;                 // row keys

#pragma unroll
    for (int st = 0; st < 8; ++st) {
        // B fragments: all 16 words contiguous at one base -> offset-immediate
        // paired LDS loads, no per-word address arithmetic.
        const _Float16* fb = &Bf[((st * 32 + lane) * BFP) * 2];
        v16h b0, b1;
#pragma unroll
        for (int v = 0; v < 8; ++v) {
            v2h p0 = *(const v2h*)&fb[2 * v];            // words 0..7  (K blk 0)
            v2h p1 = *(const v2h*)&fb[2 * (8 + v)];      // words 8..15 (K blk 1)
            b0[2 * v] = p0.x; b0[2 * v + 1] = p0.y;
            b1[2 * v] = p1.x; b1[2 * v + 1] = p1.y;
        }
        v8f acc = {};
        acc = __builtin_amdgcn_wmma_f32_16x16x32_f16(false, a0, false, b0,
                                                     (short)0, acc, false, false);
        acc = __builtin_amdgcn_wmma_f32_16x16x32_f16(false, a1, false, b1,
                                                     (short)0, acc, false, false);

        // C/D layout: VGPR r -> (M = r + 8*hlf, N = l15)
        // Integer-domain argmax: bias dots positive, one bitop3 per key,
        // v_max_u32 reductions (no float canonicalize, no cndmask chains).
        unsigned vb[8];
#pragma unroll
        for (int r = 0; r < 8; ++r)
            vb[r] = __float_as_uint(acc[r] + 2.0f);      // biased in [1,3] > 0

        const unsigned e7n = e7nBase - (unsigned)(st * 16);
#pragma unroll
        for (int r = 0; r < 8; ++r) {                    // row keys (shared n)
            unsigned rk = (vb[r] & KMASK) | e7n;
            rbest[r] = rk > rbest[r] ? rk : rbest[r];
        }

        unsigned ck = (vb[0] & KMASK) | e7mBase;         // col key tree (per-r m)
#pragma unroll
        for (int r = 1; r < 8; ++r) {
            unsigned c = (vb[r] & KMASK) | (e7mBase - (unsigned)r);
            ck = c > ck ? c : ck;
        }
        atomicMax(&ldsCol[st * 16 + l15], ck);           // ds_max_u32
    }

    // fold row state into LDS (one ds_max_u32 per (lane, r))
#pragma unroll
    for (int r = 0; r < 8; ++r)
        atomicMax(&ldsRow[mstrip + hlf * 8 + r], rbest[r]);
    __syncthreads();

    // widen 32-bit local keys to global u64 keys (hi = truncated biased bits,
    // lo = full-range enc index) and merge across workgroups.
    if (t < 128) {
        unsigned k  = ldsRow[t];
        unsigned e7 = k & 0x7Fu;                 // 127 - n_local
        unsigned long long key64 =
            ((unsigned long long)(k & KMASK) << 32) |
            (unsigned long long)(0xFFFFFFFFu - (unsigned)(nt * TILE) - 127u + e7);
        atomicMax(&rowbest[(size_t)b * MQ + mt * TILE + t], key64);
    } else {
        unsigned c  = t - 128;
        unsigned k  = ldsCol[c];
        unsigned e7 = k & 0x7Fu;                 // 127 - m_local
        unsigned long long key64 =
            ((unsigned long long)(k & KMASK) << 32) |
            (unsigned long long)(0xFFFFFFFFu - (unsigned)(mt * TILE) - 127u + e7);
        atomicMax(&colbest[(size_t)b * NQ + nt * TILE + c], key64);
    }
}

// ---- finalize: mutual-NN mask, scores, scatter ------------------------------
__global__ void matcher_finalize(const unsigned long long* __restrict__ rowbest,
                                 const unsigned long long* __restrict__ colbest,
                                 float* __restrict__ out) {
    int idx = blockIdx.x * blockDim.x + threadIdx.x;   // b*M + m
    if (idx >= BQ * MQ) return;
    int b = idx / MQ, m = idx % MQ;
    unsigned long long rk = rowbest[idx];
    int   n   = (int)(0xFFFFFFFFu - (unsigned)rk);
    float dot = __uint_as_float((unsigned)(rk >> 32)) - 2.0f;
    unsigned long long ck = colbest[(size_t)b * NQ + n];
    int mprime = (int)(0xFFFFFFFFu - (unsigned)ck);
    bool mut   = (mprime == m);
    float q     = 1.0f - dot;
    float dist  = 1.414213f * sqrtf(q > 1e-6f ? q : 1e-6f);
    float score = 1.0f / (1.0f + dist);
    const size_t BM = (size_t)BQ * MQ, BN = (size_t)BQ * NQ;
    out[idx]      = mut ? (float)n : -1.0f;            // m0
    out[BM + idx] = mut ? score : 0.0f;                // ms0
    if (mut) {                                         // mutual matches unique
        out[2 * BM + (size_t)b * NQ + n]      = (float)m;   // m1
        out[2 * BM + BN + (size_t)b * NQ + n] = score;      // ms1
    }
}

extern "C" void kernel_launch(void* const* d_in, const int* in_sizes, int n_in,
                              void* d_out, int out_size, void* d_ws, size_t ws_size,
                              hipStream_t stream) {
    const float* desc0 = (const float*)d_in[1];   // [B,M,D]
    const float* desc1 = (const float*)d_in[3];   // [B,N,D]
    float* out = (float*)d_out;

    unsigned long long* rowbest = (unsigned long long*)d_ws;        // B*M u64
    unsigned long long* colbest = rowbest + (size_t)BQ * MQ;        // B*N u64

    matcher_init<<<(BQ * MQ + 255) / 256, 256, 0, stream>>>(rowbest, colbest, out);

    dim3 grid(NQ / TILE, MQ / TILE, BQ);
    matcher_gemm_argmax<<<grid, 256, 0, stream>>>(desc0, desc1, rowbest, colbest);

    matcher_finalize<<<(BQ * MQ + 255) / 256, 256, 0, stream>>>(rowbest, colbest, out);
}
